// WindowAttention_84061099917892
// MI455X (gfx1250) — compile-verified, matrix operand order checked by reading, not run on hardware
//
#include <hip/hip_runtime.h>
#include <hip/hip_bf16.h>

typedef __attribute__((ext_vector_type(16))) _Float16 v16h;
typedef __attribute__((ext_vector_type(8)))  float    v8f;

#define NWIN 256         // tokens per window
#define HD   32          // head dim
#define NH   6           // heads
#define BH   1536        // B*H
#define TKI  128         // top-k in
#define TKO  64          // top-k out
#define CDIM 192         // channels
#define QSCALE 0.17677669529663687f   // 32^-0.5
#define PFA_EPS 1e-20f

// ---------------------------------------------------------------------------
// CDNA5 async direct-to-LDS copy (GLOBAL_LOAD_ASYNC_TO_LDS_B128, GVS mode):
// 16 bytes per lane, SGPR base + per-lane byte offset, LDS dest address in a
// VGPR.  Tracked by ASYNCcnt -> s_wait_asynccnt before the barrier.
// ---------------------------------------------------------------------------
__device__ __forceinline__ void async_copy16(unsigned lds_byte,
                                             const void* gbase,
                                             unsigned gbyte_off) {
  asm volatile("global_load_async_to_lds_b128 %0, %1, %2"
               :: "v"(lds_byte), "v"(gbyte_off), "s"(gbase)
               : "memory");
}
__device__ __forceinline__ void wait_async0() {
  asm volatile("s_wait_asynccnt 0x0" ::: "memory");
}
__device__ __forceinline__ unsigned lds_off(const void* p) {
  // generic AS pointer to __shared__: low 32 bits == LDS byte offset
  return (unsigned)(unsigned long long)p;
}

// ---------------------------------------------------------------------------
// 16x16x32 f16 WMMA fragment loader.
// A-matrix ISA layout (16-bit, 16x32 MxK): lane L holds row i = L&15,
// half = L>>4; VGPR r in [0,3] covers K = 2r + 8*half (+1); r in [4,7] covers
// K = 16 + 2(r-4) + 8*half (+1).  B (32x16 KxN) is loaded with the symmetric
// pattern: lane L holds column n = L&15, same K positions (B[k][n] = src[n][k]).
// `p` points at the tile origin of a row-major [16 x ld] f16 block.
// ---------------------------------------------------------------------------
__device__ __forceinline__ v16h ld_frag16(const _Float16* p, int ld, int lane) {
  const int i = lane & 15;
  const int hf = lane >> 4;
  v16h f;
#pragma unroll
  for (int r = 0; r < 8; ++r) {
    const int k0 = ((r < 4) ? (r << 1) : (16 + ((r - 4) << 1))) + (hf << 3);
    f[2 * r]     = p[i * ld + k0];
    f[2 * r + 1] = p[i * ld + k0 + 1];
  }
  return f;
}

// ---------------------------------------------------------------------------
// Kernel 1: stage q (pre-scaled), k, v as f16 in per-head layout, and proj_w f16
// ---------------------------------------------------------------------------
__global__ void wa_pack_kernel(const float* __restrict__ qkvp,
                               const float* __restrict__ proj_w,
                               _Float16* __restrict__ q16,
                               _Float16* __restrict__ k16,
                               _Float16* __restrict__ v16,
                               _Float16* __restrict__ w16) {
  const size_t i = (size_t)blockIdx.x * blockDim.x + threadIdx.x;
  const size_t total = (size_t)BH * NWIN * HD;   // 12,582,912
  if (i < total) {
    const size_t bh = i >> 13;               // / (NWIN*HD)
    const int b = (int)(bh / NH);
    const int h = (int)(bh % NH);
    const int n = (int)((i >> 5) & (NWIN - 1));
    const int d = (int)(i & (HD - 1));
    const size_t src = ((size_t)b * NWIN + n) * (4 * CDIM) + h * HD + d;
    q16[i] = (_Float16)(qkvp[src] * QSCALE);
    k16[i] = (_Float16)(qkvp[src + CDIM]);
    v16[i] = (_Float16)(qkvp[src + 2 * CDIM]);
  }
  if (i < (size_t)CDIM * CDIM) w16[i] = (_Float16)proj_w[i];
}

// ---------------------------------------------------------------------------
// Kernel 2: fused attention per (b,h).
// LDS: qs/ks/vs (3 * 16KB f16, staged via async-to-LDS) + Ssc (16x256 f32)
//      + sort pairs (16KB) = 80KB dynamic LDS out of the 320KB WGP pool.
// ---------------------------------------------------------------------------
__global__ void wa_attn_kernel(const _Float16* __restrict__ q16,
                               const _Float16* __restrict__ k16,
                               const _Float16* __restrict__ v16,
                               const float* __restrict__ pfa_vals,
                               const int*   __restrict__ pfa_idx,
                               const int*   __restrict__ rpi,
                               const float* __restrict__ bias_table,
                               const float* __restrict__ qkvp,
                               _Float16* __restrict__ xh,
                               float* __restrict__ out_vals,
                               float* __restrict__ out_idx) {
  extern __shared__ __align__(16) char smem[];
  _Float16* qs = (_Float16*)smem;                      // 8192 halfs
  _Float16* ks = qs + NWIN * HD;                       // 8192 halfs
  _Float16* vs = ks + NWIN * HD;                       // 8192 halfs
  float*   Ssc = (float*)(vs + NWIN * HD);             // 16*256 f32
  float*   srtv = Ssc + 16 * NWIN;                     // 16*128 f32
  int*     srts = (int*)(srtv + 16 * TKI);             // 16*128 i32

  const int tid  = threadIdx.x;
  const int wave = tid >> 5;
  const int lane = tid & 31;
  const int bh = blockIdx.x;
  const int b  = bh / NH;
  const int h  = bh % NH;

  // ---- async direct-to-LDS staging of q/k/v tiles (16KB each) ----
  {
    const void* gq = (const void*)(q16 + (size_t)bh * NWIN * HD);
    const void* gk = (const void*)(k16 + (size_t)bh * NWIN * HD);
    const void* gv = (const void*)(v16 + (size_t)bh * NWIN * HD);
    const unsigned qsb = lds_off(qs), ksb = lds_off(ks), vsb = lds_off(vs);
#pragma unroll
    for (int u = 0; u < 4; ++u) {                 // 1024 x 16B chunks / 256 thr
      const unsigned off = (unsigned)(tid + u * 256) * 16u;
      async_copy16(qsb + off, gq, off);
      async_copy16(ksb + off, gk, off);
      async_copy16(vsb + off, gv, off);
    }
    wait_async0();
  }
  __syncthreads();

  for (int r0 = 0; r0 < NWIN; r0 += 16) {
    // prefetch next chunk's gathered streams while we do WMMA on this one
    if (r0 + 16 < NWIN) {
      const int prow = r0 + 16 + wave;
      __builtin_prefetch(pfa_idx  + ((size_t)bh * NWIN + prow) * TKI + lane * 4, 0, 0);
      __builtin_prefetch(pfa_vals + ((size_t)bh * NWIN + prow) * TKI + lane * 4, 0, 0);
    }

    // ---- S chunk (16 x 256) via v_wmma_f32_16x16x32_f16: one WMMA per tile ----
    {
      const v16h a = ld_frag16(qs + r0 * HD, HD, lane);
#pragma unroll
      for (int ntl = 0; ntl < 2; ++ntl) {
        const int nt = wave + ntl * 8;
        const v16h bfr = ld_frag16(ks + nt * 16 * HD, HD, lane);
        v8f c = {};
        c = __builtin_amdgcn_wmma_f32_16x16x32_f16(
            false, a, false, bfr, (short)0, c, false, false);
        const int col = lane & 15, hf = lane >> 4;
#pragma unroll
        for (int r = 0; r < 8; ++r)
          Ssc[(r + 8 * hf) * NWIN + nt * 16 + col] = c[r];
      }
    }
    __syncthreads();

    // ---- gather + bias + softmax + PFA reweight (one row per wave, 2 passes) ----
#pragma unroll
    for (int pass = 0; pass < 2; ++pass) {
      const int rl = wave + pass * 8;
      const int row = r0 + rl;
      const int*   idxp = pfa_idx  + ((size_t)bh * NWIN + row) * TKI;
      const float* pvp  = pfa_vals + ((size_t)bh * NWIN + row) * TKI;
      float sv[4];
      float mx = -3.0e38f;
#pragma unroll
      for (int u = 0; u < 4; ++u) {
        const int j = lane + u * 32;
        const int m = idxp[j];
        const float s = Ssc[rl * NWIN + m] + bias_table[rpi[row * NWIN + m] * NH + h];
        sv[u] = s;
        mx = fmaxf(mx, s);
      }
#pragma unroll
      for (int o = 16; o > 0; o >>= 1) mx = fmaxf(mx, __shfl_xor(mx, o, 32));
      float sum = 0.f;
#pragma unroll
      for (int u = 0; u < 4; ++u) { sv[u] = __expf(sv[u] - mx); sum += sv[u]; }
#pragma unroll
      for (int o = 16; o > 0; o >>= 1) sum += __shfl_xor(sum, o, 32);
      const float inv = 1.0f / sum;
      float sum2 = 0.f;
#pragma unroll
      for (int u = 0; u < 4; ++u) {
        sv[u] = sv[u] * inv * pvp[lane + u * 32];
        sum2 += sv[u];
      }
#pragma unroll
      for (int o = 16; o > 0; o >>= 1) sum2 += __shfl_xor(sum2, o, 32);
      const float inv2 = 1.0f / (sum2 + PFA_EPS);
#pragma unroll
      for (int u = 0; u < 4; ++u) {
        const int j = lane + u * 32;
        srtv[rl * TKI + j] = (sv[u] + PFA_EPS) * inv2;
        srts[rl * TKI + j] = j;
      }
    }
    __syncthreads();

    // ---- bitonic sort (descending) of 16 rows x 128 (val,slot) pairs ----
    for (int k = 2; k <= TKI; k <<= 1) {
      for (int j = k >> 1; j > 0; j >>= 1) {
#pragma unroll 4
        for (int t = tid; t < 16 * (TKI / 2); t += 256) {
          const int rl = t >> 6;
          const int c  = t & 63;
          const int i0 = ((c & ~(j - 1)) << 1) | (c & (j - 1));
          const int i1 = i0 | j;
          const int base = rl * TKI;
          const float v0 = srtv[base + i0], v1 = srtv[base + i1];
          const bool desc = ((i0 & k) == 0);
          if (desc ? (v0 < v1) : (v0 > v1)) {
            srtv[base + i0] = v1; srtv[base + i1] = v0;
            const int s0 = srts[base + i0];
            srts[base + i0] = srts[base + i1]; srts[base + i1] = s0;
          }
        }
        __syncthreads();
      }
    }

    // ---- emit top-64 + sparse AV + lepe, stage x as f16 (one row per wave) ----
#pragma unroll
    for (int pass = 0; pass < 2; ++pass) {
      const int rl = wave + pass * 8;
      const int row = r0 + rl;
      const int* idxp = pfa_idx + ((size_t)bh * NWIN + row) * TKI;
      const size_t ob = ((size_t)bh * NWIN + row) * TKO;
#pragma unroll
      for (int u = 0; u < 2; ++u) {
        const int jj = lane + u * 32;
        out_vals[ob + jj] = srtv[rl * TKI + jj];
        out_idx[ob + jj]  = (float)idxp[srts[rl * TKI + jj]];
      }
      float acc = 0.f;                       // lane == head-dim index (HD==32)
#pragma unroll 4
      for (int jj = 0; jj < TKO; ++jj) {
        const float tv = srtv[rl * TKI + jj];
        const int m = idxp[srts[rl * TKI + jj]];
        acc += tv * (float)vs[m * HD + lane];
      }
      const float lep =
          qkvp[((size_t)b * NWIN + row) * (4 * CDIM) + 3 * CDIM + h * HD + lane];
      xh[((size_t)b * NWIN + row) * CDIM + h * HD + lane] = (_Float16)(acc + lep);
    }
    __syncthreads();
  }
}

// ---------------------------------------------------------------------------
// Kernel 3: out = x @ W^T + b per batch window (256x192 @ 192x192, K chained
// through the WMMA C operand).  LDS: x_b 96KB + W 72KB = 168KB, staged with
// async direct-to-LDS loads.
// ---------------------------------------------------------------------------
__global__ void wa_proj_kernel(const _Float16* __restrict__ xh,
                               const _Float16* __restrict__ w16,
                               const float* __restrict__ proj_b,
                               float* __restrict__ outx) {
  extern __shared__ __align__(16) char smem[];
  _Float16* xs  = (_Float16*)smem;                 // 256*192 halfs
  _Float16* wsm = xs + NWIN * CDIM;                // 192*192 halfs

  const int tid = threadIdx.x, wave = tid >> 5, lane = tid & 31;
  const int b = blockIdx.x;
  {
    const void* gx = (const void*)(xh + (size_t)b * NWIN * CDIM);
    const unsigned xsb = lds_off(xs);
#pragma unroll
    for (int u = 0; u < 24; ++u) {                // 6144 x 16B chunks
      const unsigned off = (unsigned)(tid + u * 256) * 16u;
      async_copy16(xsb + off, gx, off);
    }
    const unsigned wsb = lds_off(wsm);
#pragma unroll
    for (int u = 0; u < 18; ++u) {                // 4608 x 16B chunks
      const unsigned off = (unsigned)(tid + u * 256) * 16u;
      async_copy16(wsb + off, (const void*)w16, off);
    }
    wait_async0();
  }
  __syncthreads();

  for (int tile = wave; tile < 16 * (CDIM / 16); tile += 8) {
    const int mt = tile / (CDIM / 16);
    const int nt = tile % (CDIM / 16);
    v8f c = {};
#pragma unroll
    for (int kc = 0; kc < CDIM / 32; ++kc) {
      const v16h a   = ld_frag16(xs  + mt * 16 * CDIM + kc * 32, CDIM, lane);
      const v16h bfr = ld_frag16(wsm + nt * 16 * CDIM + kc * 32, CDIM, lane);
      c = __builtin_amdgcn_wmma_f32_16x16x32_f16(
          false, a, false, bfr, (short)0, c, false, false);
    }
    const int col = lane & 15, hf = lane >> 4;
#pragma unroll
    for (int r = 0; r < 8; ++r) {
      const int m  = mt * 16 + r + 8 * hf;
      const int cc = nt * 16 + col;
      outx[((size_t)b * NWIN + m) * CDIM + cc] = c[r] + proj_b[cc];
    }
  }
}

// ---------------------------------------------------------------------------
extern "C" void kernel_launch(void* const* d_in, const int* in_sizes, int n_in,
                              void* d_out, int out_size, void* d_ws, size_t ws_size,
                              hipStream_t stream) {
  const float* qkvp        = (const float*)d_in[0];
  const float* pfa_values  = (const float*)d_in[1];
  const int*   pfa_indices = (const int*)d_in[2];
  const int*   rpi         = (const int*)d_in[3];
  const float* bias_table  = (const float*)d_in[4];
  const float* proj_w      = (const float*)d_in[5];
  const float* proj_b      = (const float*)d_in[6];

  // workspace layout (f16): q | k | v | x_staged | W   (~96 MB total)
  const size_t HE = (size_t)BH * NWIN * HD;        // 12,582,912 elements
  _Float16* q16 = (_Float16*)d_ws;
  _Float16* k16 = q16 + HE;
  _Float16* v16 = k16 + HE;
  _Float16* xh  = v16 + HE;                        // (B, N, CDIM) staged x
  _Float16* w16 = xh + HE;                         // 192*192

  // output: x (B*N*CDIM f32) | new_vals (BH*N*64) | new_idx (BH*N*64, as float)
  float* outx     = (float*)d_out;
  float* out_vals = outx + (size_t)BH * NWIN * HD;       // == B*N*CDIM
  float* out_idx  = out_vals + (size_t)BH * NWIN * TKO;

  {
    const int threads = 256;
    const int blocks = (int)((HE + threads - 1) / threads);   // 49152
    wa_pack_kernel<<<blocks, threads, 0, stream>>>(qkvp, proj_w, q16, k16, v16, w16);
  }
  {
    // 3*16KB (q,k,v) + 16KB scores + 16KB sort pairs = 80KB dynamic LDS
    const size_t smem = (size_t)(3 * NWIN * HD * 2 + 16 * NWIN * 4 + 16 * TKI * 8);
    wa_attn_kernel<<<BH, 256, smem, stream>>>(q16, k16, v16, pfa_values, pfa_indices,
                                              rpi, bias_table, qkvp, xh,
                                              out_vals, out_idx);
  }
  {
    const size_t smem = (size_t)(NWIN * CDIM * 2 + CDIM * CDIM * 2);  // 168KB
    wa_proj_kernel<<<256, 256, smem, stream>>>(xh, w16, proj_b, outx);
  }
}
